// RowAttention_19602230739497
// MI455X (gfx1250) — matrix operand order repaired and checked by reading, not compiled, not run on hardware
//
#include <hip/hip_runtime.h>

// ---------------------------------------------------------------------------
// RowAttention for MI455X (gfx1250): bf16 WMMA pipeline.
//   B=8, C=512, Cq=64, H=W=128.  Per (b,h) slice:
//     qkv = Wqkv[640,512] @ x[512,128]          (K2, wmma bf16)
//     E   = Q''(128x64) @ K''(64x128)           (K3, wmma bf16)
//     A   = softmax_rows(E)
//     O   = A(128x128) @ V(128x512)             (K3, wmma bf16)
//     out = gamma * O^T + x
// Workspace layout (bytes):
//   [0      , 640KB ) wbuf : bf16 [640][512]  (Wq rows 0..63, Wk 64..127, Wv 128..639)
//   [1MB    , 17MB  ) qbuf : bf16 [BH][64][128]   row-major q block
//   [17MB   , 33MB  ) kbuf : bf16 [BH][128][64]   "Kt" = K'' stored N-major
//   [33MB   , 161MB ) vbuf : bf16 [BH][512][128]  V stored N-major ([c][w])
// ---------------------------------------------------------------------------

typedef __attribute__((ext_vector_type(16))) __bf16 v16bf;
typedef __attribute__((ext_vector_type(8)))  __bf16 v8bf;
typedef __attribute__((ext_vector_type(8)))  float  v8f;

// A-matrix fragment (16x32 bf16, row-major source, ld in elements).
// ISA layout: lane L<16 holds row M=L, K = {koff..koff+7, 16+koff..16+koff+7},
// koff = 8*(lane/16).  p must already point at (row, kb + 8*(lane>>4)).
__device__ __forceinline__ v16bf frag_a_rm(const __bf16* p) {
  v8bf lo = *(const v8bf*)p;
  v8bf hi = *(const v8bf*)(p + 16);
  v16bf f;
#pragma unroll
  for (int e = 0; e < 8; ++e) { f[e] = lo[e]; f[e + 8] = hi[e]; }
  return f;
}

// B-matrix fragment (32x16 bf16) from N-major storage Bt[n][k] (ld elements).
// ISA layout: lane L<16 holds N=L, K = {koff..koff+7, 16+koff..16+koff+7},
// koff = 8*(lane/16).  p must already point at (n, kb + 8*(lane>>4)).
__device__ __forceinline__ v16bf frag_b_nm(const __bf16* p) {
  v8bf lo = *(const v8bf*)p;
  v8bf hi = *(const v8bf*)(p + 16);
  v16bf f;
#pragma unroll
  for (int e = 0; e < 8; ++e) { f[e] = lo[e]; f[e + 8] = hi[e]; }
  return f;
}

__device__ __forceinline__ v8f wmma_bf16(v16bf a, v16bf b, v8f c) {
  return __builtin_amdgcn_wmma_f32_16x16x32_bf16(
      /*neg_a=*/false, a, /*neg_b=*/false, b,
      /*c_mod=*/(short)0, c, /*reuse_a=*/false, /*reuse_b=*/false);
}

// ---------------------------------------------------------------------------
// K1: pack Wq/Wk/Wv (f32) into one bf16 [640][512] matrix.
// ---------------------------------------------------------------------------
__global__ __launch_bounds__(256) void k1_pack_weights(
    const float* __restrict__ Wq, const float* __restrict__ Wk,
    const float* __restrict__ Wv, __bf16* __restrict__ wbuf) {
  int i = blockIdx.x * 256 + threadIdx.x;          // 0 .. 640*512
  if (i >= 640 * 512) return;
  float v;
  if (i < 64 * 512)        v = Wq[i];
  else if (i < 128 * 512)  v = Wk[i - 64 * 512];
  else                     v = Wv[i - 128 * 512];
  wbuf[i] = (__bf16)v;
}

// ---------------------------------------------------------------------------
// K2: per-(b,h) QKV GEMM: [640x512] @ [512x128] -> q/k/v staging buffers.
// ---------------------------------------------------------------------------
__global__ __launch_bounds__(256) void k2_qkv(
    const float* __restrict__ x, const __bf16* __restrict__ wqkv,
    const float* __restrict__ bq, const float* __restrict__ bk,
    const float* __restrict__ bv,
    __bf16* __restrict__ qbuf, __bf16* __restrict__ kbuf,
    __bf16* __restrict__ vbuf) {
  extern __shared__ char smem[];
  __bf16* xt = (__bf16*)smem;                      // xt[w][c], ld = 520 (padded)
  const int XT_LD = 520;

  const int bh = blockIdx.x;                       // 0..1023
  const int b = bh >> 7, h = bh & 127;
  const int tid = threadIdx.x;
  const size_t xbase = (size_t)b * 8388608 + (size_t)h * 128;

  // Load x slice [512 c][128 w] -> transpose into xt[w][c] as bf16.
  for (int idx = tid; idx < 512 * 32; idx += 256) {
    const int c = idx >> 5;
    const int w4 = (idx & 31) * 4;
    const float4 f = *(const float4*)(x + xbase + (size_t)c * 16384 + w4);
    xt[(w4 + 0) * XT_LD + c] = (__bf16)f.x;
    xt[(w4 + 1) * XT_LD + c] = (__bf16)f.y;
    xt[(w4 + 2) * XT_LD + c] = (__bf16)f.z;
    xt[(w4 + 3) * XT_LD + c] = (__bf16)f.w;
  }
  __syncthreads();

  const int lane = tid & 31, wave = tid >> 5;
  const int hf = lane >> 4, ln = lane & 15;
  const size_t qkb = (size_t)bh * 8192;
  const size_t vbb = (size_t)bh * 65536;

  for (int j = 0; j < 5; ++j) {                    // 40 M-tiles / 8 waves
    const int mt = wave + 8 * j;
    // Prefetch next j's A-panel row for this wave (L2-resident weight matrix).
    if (j < 4) {
      const __bf16* np = wqkv + (size_t)((mt + 8) * 16 + ln) * 512;
      __builtin_prefetch(np, 0, 2);
    }
    v8f acc[8] = {};
    for (int kb = 0; kb < 512; kb += 32) {
      const v16bf a =
          frag_a_rm(wqkv + (size_t)(mt * 16 + ln) * 512 + kb + 8 * hf);
      // Preload all 8 B fragments, then issue the WMMA burst back-to-back.
      v16bf bb[8];
#pragma unroll
      for (int nt = 0; nt < 8; ++nt)
        bb[nt] = frag_b_nm(xt + (nt * 16 + ln) * XT_LD + kb + 8 * hf);
#pragma unroll
      for (int nt = 0; nt < 8; ++nt)
        acc[nt] = wmma_bf16(a, bb[nt], acc[nt]);
    }
    // Epilogue: bias + scatter into q / k(Kt) / v(N-major) layouts.
#pragma unroll
    for (int nt = 0; nt < 8; ++nt) {
      const int n = nt * 16 + ln;                  // w coordinate
#pragma unroll
      for (int r = 0; r < 8; ++r) {
        const int m = mt * 16 + r + 8 * hf;        // output channel in [0,640)
        float val = acc[nt][r];
        if (m < 64) {
          val += bq[m];
          qbuf[qkb + m * 128 + n] = (__bf16)val;
        } else if (m < 128) {
          const int cq = m - 64;
          val += bk[cq];
          // Kt[n'][k'] with K''[kk,nn] = k[cq = nn%64, w = 2kk + nn/64]
          kbuf[qkb + (cq + 64 * (n & 1)) * 64 + (n >> 1)] = (__bf16)val;
        } else {
          const int c = m - 128;
          val += bv[c];
          vbuf[vbb + (size_t)c * 128 + n] = (__bf16)val;
        }
      }
    }
  }
}

// ---------------------------------------------------------------------------
// K3: per-(b,h) attention: E = Q''K'', softmax, O = A V, out = g*O + x.
// ---------------------------------------------------------------------------
__global__ __launch_bounds__(256) void k3_attn(
    const float* __restrict__ x, const __bf16* __restrict__ qbuf,
    const __bf16* __restrict__ kbuf, const __bf16* __restrict__ vbuf,
    const float* __restrict__ gamma, float* __restrict__ out) {
  extern __shared__ char smem[];
  float*  Es = (float*)smem;                       // [128][132] f32 scores
  __bf16* As = (__bf16*)(smem + 128 * 132 * 4);    // [128][136] bf16 probs
  const int ES_LD = 132, AS_LD = 136;

  const int bh = blockIdx.x;
  const int b = bh >> 7, h = bh & 127;
  const int tid = threadIdx.x, lane = tid & 31, wave = tid >> 5;
  const int hf = lane >> 4, ln = lane & 15;
  const __bf16* qb = qbuf + (size_t)bh * 8192;
  const __bf16* kt = kbuf + (size_t)bh * 8192;
  const __bf16* vb = vbuf + (size_t)bh * 65536;

  // Prefetch this wave's V panel (streams from HBM in the O-GEMM below).
  {
    const __bf16* vp = vb + (size_t)(wave * 4 * 16 + ln) * 128;
    __builtin_prefetch(vp, 0, 1);
  }

  // ---- E = Q''(128x64) @ K''(64x128); one M-tile per wave ----
  {
    const int mt = wave;
    v8f acc[8] = {};
#pragma unroll
    for (int kb = 0; kb < 64; kb += 32) {
      const v16bf a = frag_a_rm(qb + (mt * 16 + ln) * 64 + kb + 8 * hf);
      v16bf bb[8];
#pragma unroll
      for (int nt = 0; nt < 8; ++nt)
        bb[nt] = frag_b_nm(kt + (nt * 16 + ln) * 64 + kb + 8 * hf);
#pragma unroll
      for (int nt = 0; nt < 8; ++nt)
        acc[nt] = wmma_bf16(a, bb[nt], acc[nt]);
    }
#pragma unroll
    for (int nt = 0; nt < 8; ++nt)
#pragma unroll
      for (int r = 0; r < 8; ++r)
        Es[(mt * 16 + r + 8 * hf) * ES_LD + nt * 16 + ln] = acc[nt][r];
  }
  __syncthreads();

  // ---- row softmax (one thread per row) ----
  if (tid < 128) {
    float* row = Es + tid * ES_LD;
    float mx = -1e30f;
    for (int jj = 0; jj < 128; ++jj) mx = fmaxf(mx, row[jj]);
    float s = 0.f;
    for (int jj = 0; jj < 128; ++jj) {
      const float e = __expf(row[jj] - mx);
      row[jj] = e;
      s += e;
    }
    const float inv = 1.f / s;
    __bf16* arow = As + tid * AS_LD;
    for (int jj = 0; jj < 128; ++jj) arow[jj] = (__bf16)(row[jj] * inv);
  }
  __syncthreads();

  // ---- O = A(128x128) @ V(128x512); wave owns 64 N-columns ----
  const float gm = gamma[0];
  for (int mc = 0; mc < 2; ++mc) {                 // two 64-row M chunks
    v8f acc[4][4] = {};
    for (int kb = 0; kb < 128; kb += 32) {
      // Preload 4 A frags (LDS) and 4 B frags (global V panel), then burst.
      v16bf afr[4], bfr[4];
#pragma unroll
      for (int mi = 0; mi < 4; ++mi)
        afr[mi] = frag_a_rm(As + ((mc * 4 + mi) * 16 + ln) * AS_LD + kb + 8 * hf);
#pragma unroll
      for (int ni = 0; ni < 4; ++ni) {
        const int n = (wave * 4 + ni) * 16 + ln;   // channel c
        bfr[ni] = frag_b_nm(vb + (size_t)n * 128 + kb + 8 * hf);
      }
#pragma unroll
      for (int ni = 0; ni < 4; ++ni)
#pragma unroll
        for (int mi = 0; mi < 4; ++mi)
          acc[mi][ni] = wmma_bf16(afr[mi], bfr[ni], acc[mi][ni]);
    }
    // Fused epilogue: out[b,c,h,w] = gamma * O[w,c] + x[b,c,h,w]
#pragma unroll
    for (int mi = 0; mi < 4; ++mi)
#pragma unroll
      for (int ni = 0; ni < 4; ++ni) {
        const int c = (wave * 4 + ni) * 16 + ln;
        const size_t base = (size_t)b * 8388608 + (size_t)c * 16384 + (size_t)h * 128;
#pragma unroll
        for (int r = 0; r < 8; ++r) {
          const int i = (mc * 4 + mi) * 16 + r + 8 * hf;  // w coordinate
          out[base + i] = gm * acc[mi][ni][r] + x[base + i];
        }
      }
  }
}

// ---------------------------------------------------------------------------
extern "C" void kernel_launch(void* const* d_in, const int* in_sizes, int n_in,
                              void* d_out, int out_size, void* d_ws, size_t ws_size,
                              hipStream_t stream) {
  (void)in_sizes; (void)n_in; (void)out_size; (void)ws_size;
  const float* x     = (const float*)d_in[0];
  const float* Wq    = (const float*)d_in[1];
  const float* bq    = (const float*)d_in[2];
  const float* Wk    = (const float*)d_in[3];
  const float* bk    = (const float*)d_in[4];
  const float* Wv    = (const float*)d_in[5];
  const float* bv    = (const float*)d_in[6];
  const float* gamma = (const float*)d_in[7];
  float* out = (float*)d_out;

  char* ws = (char*)d_ws;
  __bf16* wbuf = (__bf16*)(ws);
  __bf16* qbuf = (__bf16*)(ws + (size_t)(1)  * 1048576);
  __bf16* kbuf = (__bf16*)(ws + (size_t)(17) * 1048576);
  __bf16* vbuf = (__bf16*)(ws + (size_t)(33) * 1048576);

  k1_pack_weights<<<1280, 256, 0, stream>>>(Wq, Wk, Wv, wbuf);

  const size_t k2_lds = 128 * 520 * sizeof(__bf16);              // 133,120 B
  k2_qkv<<<1024, 256, k2_lds, stream>>>(x, wbuf, bq, bk, bv, qbuf, kbuf, vbuf);

  const size_t k3_lds = 128 * 132 * 4 + 128 * 136 * 2;           // 102,400 B
  k3_attn<<<1024, 256, k3_lds, stream>>>(x, qbuf, kbuf, vbuf, gamma, out);
}
